// Transformer_28853590294678
// MI455X (gfx1250) — compile-verified, matrix operand order checked by reading, not run on hardware
//
#include <hip/hip_runtime.h>
#include <math.h>

typedef __attribute__((ext_vector_type(16))) _Float16 v16h;
typedef __attribute__((ext_vector_type(8)))  _Float16 v8h;
typedef __attribute__((ext_vector_type(8)))  float    v8f;

#define HEADS 8
#define DIMH 32
#define NTOK 1024            // 32*32
#define BATCH 16
#define CDIM 256
#define HID 1024
#define MROWS (BATCH * NTOK) // 16384
#define QKV_STRIDE ((size_t)BATCH * HEADS * NTOK * DIMH) // 4194304
#define LOG2E 1.44269504088896340f

#if defined(__has_builtin)
#if __has_builtin(__builtin_amdgcn_global_load_async_to_lds_b128)
#define HAVE_ASYNC 1
typedef int v4i_t __attribute__((vector_size(16)));
typedef __attribute__((address_space(1))) v4i_t* gas1_t;
typedef __attribute__((address_space(3))) v4i_t* las3_t;
#endif
#endif

// ---------------- DPP16 butterfly reductions (no DS traffic) ----------------
template <int CTRL>
__device__ __forceinline__ float dpp_movf(float v) {
    return __int_as_float(
        __builtin_amdgcn_update_dpp(0, __float_as_int(v), CTRL, 0xF, 0xF, true));
}
// single-instruction max (avoids cmp/cndmask chains and sNaN-quieting movs)
__device__ __forceinline__ float fmax2(float a, float b) {
    float d;
    asm("v_max_num_f32 %0, %1, %2" : "=v"(d) : "v"(a), "v"(b));
    return d;
}
// full max reduction over each 16-lane half: masks {1,2,7,15} = basis of GF(2)^4
__device__ __forceinline__ float red16_max(float v) {
    v = fmax2(v, dpp_movf<0x0B1>(v)); // quad_perm [1,0,3,2]  (xor 1)
    v = fmax2(v, dpp_movf<0x04E>(v)); // quad_perm [2,3,0,1]  (xor 2)
    v = fmax2(v, dpp_movf<0x141>(v)); // row_half_mirror      (xor 7)
    v = fmax2(v, dpp_movf<0x140>(v)); // row_mirror           (xor 15)
    return v;
}
// sum reduction over aligned 8-lane groups: masks {1,2,7}
__device__ __forceinline__ float red8_sum(float v) {
    v += dpp_movf<0x0B1>(v);
    v += dpp_movf<0x04E>(v);
    v += dpp_movf<0x141>(v);
    return v;
}

// ---------- fragment load helper: 16 halfs per lane, WMMA A/B layout ----------
__device__ __forceinline__ v16h ldfrag(const _Float16* base, int lhi) {
    v8h lo = *(const v8h*)(base + lhi * 8);
    v8h hi = *(const v8h*)(base + 16 + lhi * 8);
    v16h r;
#pragma unroll
    for (int j = 0; j < 8; ++j) { r[j] = lo[j]; r[8 + j] = hi[j]; }
    return r;
}

__device__ __forceinline__ v8f wmma_f16(v16h a, v16h b, v8f c) {
    return __builtin_amdgcn_wmma_f32_16x16x32_f16(false, a, false, b, (short)0, c, false, false);
}

// ---------- weight transpose + f32 -> f16 : WT[n*K+k] = W[k*N+n] ----------
__global__ void transpose_to_f16(const float* __restrict__ W, _Float16* __restrict__ WT,
                                 int K, int N) {
    int idx = blockIdx.x * blockDim.x + threadIdx.x;
    if (idx >= K * N) return;
    int n = idx / K, k = idx % K;
    WT[idx] = (_Float16)W[(size_t)k * N + n];
}

// ---- rel-pos bias gather, f32, pre-scaled by log2e, in WMMA C-fragment order ----
// addr(h,i,j) = (h<<20) + ((i>>4)*64 + (j>>4))*256 + lane*8 + r
// where lane = (j&15) + 16*((i>>3)&1), r = i&7   (matches C/D VGPR layout)
__global__ void bias_gather(const float* __restrict__ table, const int* __restrict__ relidx,
                            float* __restrict__ biasw) {
    int idx = blockIdx.x * blockDim.x + threadIdx.x; // over n*n = 1M
    if (idx >= NTOK * NTOK) return;
    int i = idx >> 10, j = idx & 1023;
    int ti = i >> 4, mr = i & 15;
    int tj = j >> 4, nc = j & 15;
    int lane = nc + ((mr >> 3) << 4);
    int r = mr & 7;
    size_t tilebase = ((size_t)(ti * 64 + tj)) * 256 + lane * 8 + r;
    int rel = relidx[idx];
#pragma unroll
    for (int h = 0; h < HEADS; ++h)
        biasw[((size_t)h << 20) + tilebase] = table[rel * HEADS + h] * LOG2E;
}

// ---------- layernorm over C: X (b,c,n) f32 -> Y (b,n,c) f16 ----------
__global__ __launch_bounds__(256)
void ln_kernel(const float* __restrict__ X, const float* __restrict__ gw,
               const float* __restrict__ gb, _Float16* __restrict__ Y) {
    __shared__ float sm[CDIM][33];
    int tid = threadIdx.x;
    int b = blockIdx.y;
    int i0 = blockIdx.x * 32;
    int il = tid & 31, cg = tid >> 5;
#pragma unroll 4
    for (int it = 0; it < 32; ++it) {
        int c = cg * 32 + it;
        sm[c][il] = X[((size_t)b * CDIM + c) * NTOK + i0 + il];
    }
    __syncthreads();
    int i_loc = tid >> 3, sub = tid & 7; // 8 threads per row (aligned 8-lane groups)
    float s = 0.f, s2 = 0.f;
#pragma unroll
    for (int cc = 0; cc < 32; ++cc) {
        float v = sm[sub * 32 + cc][i_loc];
        s += v; s2 += v * v;
    }
    s  = red8_sum(s);
    s2 = red8_sum(s2);
    float mu  = s * (1.f / CDIM);
    float var = s2 * (1.f / CDIM) - mu * mu;
    float inv = rsqrtf(var + 1e-5f);
    size_t obase = ((size_t)b * NTOK + i0 + i_loc) * CDIM;
#pragma unroll
    for (int cc = 0; cc < 32; ++cc) {
        int c = sub * 32 + cc;
        float v = (sm[c][i_loc] - mu) * inv * gw[c] + gb[c];
        Y[obase + c] = (_Float16)v;
    }
}

// ---------- generic WMMA GEMM: A (M x K) f16 row-major, BT (N x K) f16 ----------
// MODE 0: QKV  -> split heads, q pre-scaled by d^-1/2 * log2e, f16 out
// MODE 1: PROJ -> +proj_b, + residual x, f32 out in (b,c,n)
// MODE 2: FFN1 -> +b1, exact GELU, f16 out row-major (M x N)
// MODE 3: FFN2 -> +b2, + x3, f32 out (b,c,n) == d_out
template <int MODE, int K>
__global__ __launch_bounds__(256)
void gemm_wmma(const _Float16* __restrict__ A, const _Float16* __restrict__ BT,
               int N,
               const float* __restrict__ cbias,
               const float* __restrict__ resid,
               float* __restrict__ outf,
               _Float16* __restrict__ outh) {
    int lane = threadIdx.x & 31;
    int wave = threadIdx.x >> 5;
    int lhi = lane >> 4, lmod = lane & 15;
    int m0 = blockIdx.y * 128 + wave * 16;
    int n0 = blockIdx.x * 64;
    int row = m0 + lmod;

    v8f acc[4];
#pragma unroll
    for (int t = 0; t < 4; ++t) acc[t] = (v8f){0.f,0.f,0.f,0.f,0.f,0.f,0.f,0.f};

    const _Float16* arow = A + (size_t)row * K;
#pragma unroll 4
    for (int k = 0; k < K; k += 32) {
        if (k + 32 < K) __builtin_prefetch(arow + k + 32, 0, 0);
        v16h a = ldfrag(arow + k, lhi);
#pragma unroll
        for (int t = 0; t < 4; ++t) {
            int col = n0 + t * 16 + lmod;
            v16h bfr = ldfrag(BT + (size_t)col * K + k, lhi);
            acc[t] = wmma_f16(a, bfr, acc[t]);
        }
    }

#pragma unroll
    for (int t = 0; t < 4; ++t) {
        int col = n0 + t * 16 + lmod;
#pragma unroll
        for (int r = 0; r < 8; ++r) {
            float v = acc[t][r];
            int mi = m0 + r + 8 * lhi;       // global row
            int b = mi >> 10, i = mi & 1023; // batch, token
            if (MODE == 0) {
                int which = col >> 8, rem = col & 255;
                int h = rem >> 5, d = rem & 31;
                // fold softmax scale AND log2e (exp2-domain softmax) into q
                if (which == 0) v *= (0.17677669529663689f * LOG2E);
                outh[(size_t)which * QKV_STRIDE +
                     (((size_t)(b * HEADS + h) * NTOK + i) * DIMH + d)] = (_Float16)v;
            } else if (MODE == 1) {
                v += cbias[col];
                size_t addr = ((size_t)b * CDIM + col) * NTOK + i;
                outf[addr] = v + resid[addr];
            } else if (MODE == 2) {
                v += cbias[col];
                v = 0.5f * v * (1.0f + erff(v * 0.70710678118654752f));
                outh[(size_t)mi * HID + col] = (_Float16)v;
            } else { // MODE == 3
                v += cbias[col];
                size_t addr = ((size_t)b * CDIM + col) * NTOK + i;
                outf[addr] = resid[addr] + v;
            }
        }
    }
}

// ---------- flash attention: block = 4 waves x 16 rows, per (b, h, i-tile64) ----------
// scores are in the exp2 domain (q and bias pre-scaled by log2e): p = exp2(s - m)
#define JCHUNK 256
__global__ __launch_bounds__(128)
void attn_kernel(const _Float16* __restrict__ qh, const _Float16* __restrict__ kh,
                 const _Float16* __restrict__ vh, const float* __restrict__ biasw,
                 _Float16* __restrict__ outh) {
    __shared__ __align__(16) _Float16 Ksh[JCHUNK * DIMH];   // 16 KB  [j][d]
    __shared__ __align__(16) _Float16 Vt[DIMH * JCHUNK];    // 16 KB  [d][j]
    __shared__ __align__(16) _Float16 Psh[4][16 * 32];      // 4 KB   per-wave P bounce

    int b = blockIdx.z, h = blockIdx.y;
    int tid = threadIdx.x, lane = tid & 31, wave = tid >> 5;
    int lhi = lane >> 4, lmod = lane & 15;
    int ibase = blockIdx.x * 64 + wave * 16;

    const _Float16* Kg = kh + ((size_t)(b * HEADS + h) * NTOK) * DIMH;
    const _Float16* Vg = vh + ((size_t)(b * HEADS + h) * NTOK) * DIMH;

    // Q fragment (already scaled at QKV epilogue)
    v16h qf = ldfrag(qh + (((size_t)(b * HEADS + h) * NTOK + ibase + lmod)) * DIMH, lhi);

    // all-ones B fragment: makes every lane of o2 the exact row-sum of P
    v16h onesf;
#pragma unroll
    for (int j = 0; j < 16; ++j) onesf[j] = (_Float16)1.0f;

    v8f o0 = (v8f){0,0,0,0,0,0,0,0}, o1 = (v8f){0,0,0,0,0,0,0,0};
    v8f o2 = (v8f){0,0,0,0,0,0,0,0}; // running softmax denominators (per row)
    float mrow[8];
#pragma unroll
    for (int r = 0; r < 8; ++r) mrow[r] = -1e30f;

    // per-lane base into swizzled f32 bias fragments for this (h, i-tile)
    const float* bias_lane =
        biasw + ((size_t)h << 20) + (size_t)(ibase >> 4) * 64 * 256 + (size_t)lane * 8;

    for (int jc = 0; jc < NTOK; jc += JCHUNK) {
        __syncthreads(); // previous chunk fully consumed
#ifdef HAVE_ASYNC
        { // K chunk via async global->LDS DMA (16 KB, 1024 x b128)
            const char* src = (const char*)(Kg + (size_t)jc * DIMH);
            char* dst = (char*)Ksh;
            for (int u = tid; u < JCHUNK * DIMH / 8; u += 128)
                __builtin_amdgcn_global_load_async_to_lds_b128(
                    (gas1_t)(src + (size_t)u * 16), (las3_t)(dst + (size_t)u * 16), 0, 0);
        }
#else
        { // stage K chunk (row-major) — 1024 x 16B
            const uint4* src = (const uint4*)(Kg + (size_t)jc * DIMH);
            uint4* dst = (uint4*)Ksh;
            for (int u = tid; u < JCHUNK * DIMH / 8; u += 128) dst[u] = src[u];
        }
#endif
        { // stage V chunk transposed: Vt[d][j]
            for (int u = tid; u < JCHUNK * DIMH; u += 128) {
                int j = u >> 5, d = u & 31;
                Vt[d * JCHUNK + j] = Vg[(size_t)(jc + j) * DIMH + d];
            }
        }
#ifdef HAVE_ASYNC
        asm volatile("s_wait_asynccnt 0" ::: "memory");
#endif
        __syncthreads();

        for (int j0 = 0; j0 < JCHUNK; j0 += 32) {
            // K^T B-fragments: column j fixed per lane, contiguous over d
            v16h kb0 = ldfrag(Ksh + (j0 + lmod) * DIMH, lhi);
            v16h kb1 = ldfrag(Ksh + (j0 + 16 + lmod) * DIMH, lhi);
            // bias tiles straight into the C operand: one v8f per tile per lane
            int tj = (jc + j0) >> 4;
            v8f c0 = *(const v8f*)(bias_lane + (size_t)tj * 256);
            v8f c1 = *(const v8f*)(bias_lane + (size_t)(tj + 1) * 256);

            v8f s0 = wmma_f16(qf, kb0, c0);
            v8f s1 = wmma_f16(qf, kb1, c1);

            // online softmax: DPP max reduction; row-sum comes from the ones-WMMA
#pragma unroll
            for (int r = 0; r < 8; ++r) {
                float mx = red16_max(fmax2(s0[r], s1[r]));
                float nm = fmax2(mrow[r], mx);
                float alpha = exp2f(mrow[r] - nm);   // v_exp_f32 is native base-2
                mrow[r] = nm;
                s0[r] = exp2f(s0[r] - nm);
                s1[r] = exp2f(s1[r] - nm);
                o0[r] *= alpha; o1[r] *= alpha; o2[r] *= alpha;
            }

            // bounce P through LDS: C-layout -> A-layout (wave-local, DS in-order)
            _Float16* P = &Psh[wave][0];
#pragma unroll
            for (int r = 0; r < 8; ++r) {
                int mi = r + 8 * lhi;
                P[mi * 32 + lmod]      = (_Float16)s0[r];
                P[mi * 32 + 16 + lmod] = (_Float16)s1[r];
            }
            asm volatile("s_wait_dscnt 0" ::: "memory");
            v16h pf  = ldfrag(P + lmod * 32, lhi);
            v16h vb0 = ldfrag(Vt + (size_t)lmod * JCHUNK + j0, lhi);        // d = 0..15
            v16h vb1 = ldfrag(Vt + (size_t)(16 + lmod) * JCHUNK + j0, lhi); // d = 16..31
            o0 = wmma_f16(pf, vb0, o0);
            o1 = wmma_f16(pf, vb1, o1);
            o2 = wmma_f16(pf, onesf, o2); // row-sums of f16 P, replicated to all lanes
        }
    }

    // normalize and store (b, i, h*32+d) f16 for the proj GEMM
#pragma unroll
    for (int r = 0; r < 8; ++r) {
        float inv = 1.f / o2[r];
        int i = ibase + r + 8 * lhi;
        size_t base = ((size_t)b * NTOK + i) * CDIM + h * DIMH;
        outh[base + lmod]      = (_Float16)(o0[r] * inv);
        outh[base + 16 + lmod] = (_Float16)(o1[r] * inv);
    }
}

// ------------------------------- launcher -------------------------------
extern "C" void kernel_launch(void* const* d_in, const int* in_sizes, int n_in,
                              void* d_out, int out_size, void* d_ws, size_t ws_size,
                              hipStream_t stream) {
    const float* x      = (const float*)d_in[0];
    const float* qkv_w  = (const float*)d_in[1];
    const float* proj_w = (const float*)d_in[2];
    const float* proj_b = (const float*)d_in[3];
    const float* ffn_w1 = (const float*)d_in[4];
    const float* ffn_b1 = (const float*)d_in[5];
    const float* ffn_w2 = (const float*)d_in[6];
    const float* ffn_b2 = (const float*)d_in[7];
    const float* n1w = (const float*)d_in[8];
    const float* n1b = (const float*)d_in[9];
    const float* n2w = (const float*)d_in[10];
    const float* n2b = (const float*)d_in[11];
    const float* bias_table = (const float*)d_in[12];
    const int*   rel_index  = (const int*)d_in[13];
    float* out = (float*)d_out;

    char* wsp = (char*)d_ws;
    size_t off = 0;
    auto alloc = [&](size_t bytes) -> void* {
        void* p = wsp + off;
        off += (bytes + 255) & ~(size_t)255;
        return p;
    };
    _Float16* x1h    = (_Float16*)alloc((size_t)MROWS * CDIM * 2);
    _Float16* qkvh   = (_Float16*)alloc(3 * QKV_STRIDE * 2);
    _Float16* attnh  = (_Float16*)alloc((size_t)MROWS * CDIM * 2);
    _Float16* x4h    = (_Float16*)alloc((size_t)MROWS * CDIM * 2);
    _Float16* ffn1h  = (_Float16*)alloc((size_t)MROWS * HID * 2);
    _Float16* wqkvT  = (_Float16*)alloc((size_t)768 * 256 * 2);
    _Float16* wprojT = (_Float16*)alloc((size_t)256 * 256 * 2);
    _Float16* w1T    = (_Float16*)alloc((size_t)1024 * 256 * 2);
    _Float16* w2T    = (_Float16*)alloc((size_t)256 * 1024 * 2);
    float*    biasw  = (float*)alloc((size_t)HEADS * NTOK * NTOK * 4);
    float*    x3     = (float*)alloc((size_t)MROWS * CDIM * 4);
    (void)ws_size; (void)n_in; (void)in_sizes; (void)out_size;

    // weight precompute (transpose + f16)
    transpose_to_f16<<<(256 * 768 + 255) / 256, 256, 0, stream>>>(qkv_w, wqkvT, 256, 768);
    transpose_to_f16<<<(256 * 256 + 255) / 256, 256, 0, stream>>>(proj_w, wprojT, 256, 256);
    transpose_to_f16<<<(256 * 1024 + 255) / 256, 256, 0, stream>>>(ffn_w1, w1T, 256, 1024);
    transpose_to_f16<<<(1024 * 256 + 255) / 256, 256, 0, stream>>>(ffn_w2, w2T, 1024, 256);
    // relative-position bias, pre-swizzled to C-fragment order (reused by all batches)
    bias_gather<<<(NTOK * NTOK) / 256, 256, 0, stream>>>(bias_table, rel_index, biasw);

    // LN1: (b,c,n) f32 -> (b,n,c) f16
    ln_kernel<<<dim3(32, 16), 256, 0, stream>>>(x, n1w, n1b, x1h);
    // QKV GEMM 16384 x 256 x 768 -> head-split q,k,v (q pre-scaled)
    gemm_wmma<0, 256><<<dim3(12, 128), 256, 0, stream>>>(x1h, wqkvT, 768,
                                                         nullptr, nullptr, nullptr, qkvh);
    // flash attention with fused relative-position bias
    attn_kernel<<<dim3(16, 8, 16), 128, 0, stream>>>(qkvh, qkvh + QKV_STRIDE,
                                                     qkvh + 2 * QKV_STRIDE, biasw, attnh);
    // proj + bias + residual -> x3 (b,c,n) f32
    gemm_wmma<1, 256><<<dim3(4, 128), 256, 0, stream>>>(attnh, wprojT, 256,
                                                        proj_b, x, x3, nullptr);
    // LN2
    ln_kernel<<<dim3(32, 16), 256, 0, stream>>>(x3, n2w, n2b, x4h);
    // FFN1 + exact GELU
    gemm_wmma<2, 256><<<dim3(16, 128), 256, 0, stream>>>(x4h, w1T, 1024,
                                                         ffn_b1, nullptr, nullptr, ffn1h);
    // FFN2 + final residual -> d_out (b,c,h,w) f32
    gemm_wmma<3, 1024><<<dim3(4, 128), 256, 0, stream>>>(ffn1h, w2T, 256,
                                                         ffn_b2, x3, out, nullptr);
}
